// AttentionLayer_31610959299130
// MI455X (gfx1250) — matrix operand order repaired and checked by reading, not compiled, not run on hardware
//
#include <hip/hip_runtime.h>
#include <hip/hip_bf16.h>

// ---------------------------------------------------------------------------
// MixHop GCN block for MI455X (gfx1250, wave32, WMMA).
//   prop:  one wave32 per edge, float4/lane gather + global_atomic_add_f32
//   gemm:  one wave32 per 16x16 tile, v_wmma_f32_16x16x32_f16, f32 accum.
//          A is f16 row-major (fragment = 2x b128/lane), W is pre-transposed
//          f16 [N][K] (fragment = 1x 32B vector load/lane) -> inner loop is
//          4 vector loads + 1 WMMA per K=32 step, zero conversions.
// ---------------------------------------------------------------------------

typedef __attribute__((ext_vector_type(16))) _Float16 v16h;
typedef __attribute__((ext_vector_type(8)))  _Float16 v8h;
typedef __attribute__((ext_vector_type(8)))  float    v8f;

#define N_NODES 50000
#define E_DIM   128
#define H_DIM   512
#define C3      (3 * E_DIM)

// ---------------- degree / normalization ----------------------------------

__global__ void deg_kernel(const int* __restrict__ dst, float* __restrict__ deg, int E) {
    int e = blockIdx.x * blockDim.x + threadIdx.x;
    if (e < E) atomicAdd(&deg[dst[e]], 1.0f);
}

__global__ void dinv_kernel(const float* __restrict__ deg, float* __restrict__ dinv, int n) {
    int i = blockIdx.x * blockDim.x + threadIdx.x;
    if (i < n) {
        float d = deg[i];
        dinv[i] = (d > 0.0f) ? rsqrtf(d) : 0.0f;
    }
}

// ---------------- normalized scatter-add propagation -----------------------
// one wave32 per edge; lane l handles channels [4l, 4l+3] (float4 = 128 ch)

__global__ void prop_kernel(const int* __restrict__ src, const int* __restrict__ dst,
                            const float* __restrict__ dinv,
                            const float* __restrict__ in, float* __restrict__ out, int E) {
    int gt   = blockIdx.x * blockDim.x + threadIdx.x;
    int e    = gt >> 5;
    int lane = gt & 31;
    if (e >= E) return;
    // stream the index arrays ahead through L2 (global_prefetch_b8)
    if (e + 8192 < E) {
        __builtin_prefetch(src + e + 8192, 0, 1);
        __builtin_prefetch(dst + e + 8192, 0, 1);
    }
    int s = src[e], d = dst[e];
    float w = dinv[s] * dinv[d];
    const float4 v = ((const float4*)(in + (size_t)s * E_DIM))[lane];
    float* o = out + (size_t)d * E_DIM + lane * 4;
    atomicAdd(o + 0, w * v.x);
    atomicAdd(o + 1, w * v.y);
    atomicAdd(o + 2, w * v.z);
    atomicAdd(o + 3, w * v.w);
}

// ---------------- f16 staging ----------------------------------------------

// elementwise f32 -> f16, 4 elements/thread (n multiple of 4)
__global__ void cvt_f16x4(const float* __restrict__ in, _Float16* __restrict__ out, size_t n4) {
    size_t i = (size_t)blockIdx.x * blockDim.x + threadIdx.x;
    if (i >= n4) return;
    float4 v = ((const float4*)in)[i];
    _Float16* o = out + i * 4;
    o[0] = (_Float16)v.x; o[1] = (_Float16)v.y;
    o[2] = (_Float16)v.z; o[3] = (_Float16)v.w;
}

// W[K][N] f32 -> Wt[N][K] f16 (weights are small; done once per launch)
__global__ void transpose_f16(const float* __restrict__ W, _Float16* __restrict__ Wt,
                              int K, int N) {
    int idx = blockIdx.x * blockDim.x + threadIdx.x;
    if (idx >= K * N) return;
    int k = idx / N, n = idx - k * N;
    Wt[(size_t)n * K + k] = (_Float16)W[idx];
}

// ---------------- WMMA GEMM -------------------------------------------------
// C[M,Nc](+col_off) = op(A[M,K]_f16 @ Wt[Nc,K]_f16^T + bias)
// One wave per 16x16 output tile.  Fragment loads use the same contiguous k order in A and B:
//   A frag: lane(M=l%16,g=l/16): halves [k0+16g .. k0+16g+15] contiguous in A row (2x 8-elem loads)
//   B frag: lane(N=l%16,g=l/16): halves [k0+16g .. k0+16g+15] contiguous in Wt row
//   C/D:    VGPR r -> row r+8g, col l%16

template <bool RELU, bool OUT_F16, bool SKIP>
__global__ void wmma_gemm(const _Float16* __restrict__ A, int lda,
                          const _Float16* __restrict__ Wt, int ldw,
                          const float* __restrict__ bias,
                          const _Float16* __restrict__ skip, int skip_ld,
                          void* __restrict__ C_, int ldc, int col_off,
                          int M, int K, int tiles_n) {
    int wave = (blockIdx.x * blockDim.x + threadIdx.x) >> 5;
    int lane = threadIdx.x & 31;
    int tiles_m = M >> 4;                  // M is a multiple of 16 here
    if (wave >= tiles_m * tiles_n) return; // uniform whole-wave exit (EXEC all-1 for WMMA)

    int tm = wave / tiles_n;
    int tn = wave - tm * tiles_n;
    int row_base = tm << 4;
    int col_base = tn << 4;
    int m = lane & 15;
    int g = lane >> 4;

    const _Float16* Arow = A  + (size_t)(row_base + m) * lda + 16 * g;
    const _Float16* Brow = Wt + (size_t)(col_base + m) * ldw + 16 * g;

    v8f c = {};
    for (int k0 = 0; k0 < K; k0 += 32) {
        v8h a0 = *(const v8h*)(Arow + k0);        // K = k0+16g  .. +7
        v8h a1 = *(const v8h*)(Arow + k0 + 8);    // K = k0+16g+8.. +7
        v16h a = __builtin_shufflevector(a0, a1, 0, 1, 2, 3, 4, 5, 6, 7,
                                         8, 9, 10, 11, 12, 13, 14, 15);
        v16h b = *(const v16h*)(Brow + k0);       // K = k0+16g .. +15
        c = __builtin_amdgcn_wmma_f32_16x16x32_f16(false, a, false, b,
                                                   (short)0, c, false, false);
    }

    int col = col_base + m;
#pragma unroll
    for (int r = 0; r < 8; ++r) {
        int row = row_base + r + 8 * g;
        float v = c[r] + bias[col];
        if (RELU) v = fmaxf(v, 0.0f);
        if (SKIP) v += (float)skip[(size_t)row * skip_ld + col];
        if (OUT_F16)
            ((_Float16*)C_)[(size_t)row * ldc + col + col_off] = (_Float16)v;
        else
            ((float*)C_)[(size_t)row * ldc + col + col_off] = v;
    }
}

// ---------------- batchnorm (training-mode, biased var) --------------------

__global__ void bn_stats(const float* __restrict__ h, int nrows, int C,
                         float* __restrict__ sums) {  // sums[0:C]=sum, [C:2C]=sumsq
    int c = threadIdx.x;                              // blockDim.x == C
    float s = 0.0f, s2 = 0.0f;
    for (int r = blockIdx.x; r < nrows; r += gridDim.x) {
        float v = h[(size_t)r * C + c];
        s += v;
        s2 += v * v;
    }
    atomicAdd(&sums[c], s);
    atomicAdd(&sums[C + c], s2);
}

__global__ void bn_finalize(const float* __restrict__ sums,
                            const float* __restrict__ gamma,
                            const float* __restrict__ beta,
                            float* __restrict__ ss, int C, float inv_n) {
    int c = blockIdx.x * blockDim.x + threadIdx.x;
    if (c >= C) return;
    float mu  = sums[c] * inv_n;
    float var = sums[C + c] * inv_n - mu * mu;
    float sc  = gamma[c] * rsqrtf(var + 1e-5f);
    ss[c]     = sc;
    ss[C + c] = beta[c] - mu * sc;
}

// BN apply fused with f16 conversion (for the FFN input)
__global__ void bn_apply_f16(const float* __restrict__ h, const float* __restrict__ ss,
                             _Float16* __restrict__ out, int C, size_t total) {
    size_t i = (size_t)blockIdx.x * blockDim.x + threadIdx.x;
    if (i >= total) return;
    int c = (int)(i % C);
    out[i] = (_Float16)(h[i] * ss[c] + ss[C + c]);
}

// BN apply in place, f32 (final output)
__global__ void bn_apply(float* __restrict__ h, const float* __restrict__ ss,
                         int C, size_t total) {
    size_t i = (size_t)blockIdx.x * blockDim.x + threadIdx.x;
    if (i >= total) return;
    int c = (int)(i % C);
    h[i] = h[i] * ss[c] + ss[C + c];
}

// ---------------- host-side orchestration ----------------------------------

extern "C" void kernel_launch(void* const* d_in, const int* in_sizes, int n_in,
                              void* d_out, int out_size, void* d_ws, size_t ws_size,
                              hipStream_t stream) {
    const float* x      = (const float*)d_in[0];
    const int*   ei     = (const int*)d_in[1];
    const float* W0     = (const float*)d_in[2];
    const float* b0     = (const float*)d_in[3];
    const float* W1     = (const float*)d_in[4];
    const float* b1     = (const float*)d_in[5];
    const float* W2     = (const float*)d_in[6];
    const float* b2     = (const float*)d_in[7];
    const float* gamma1 = (const float*)d_in[8];
    const float* beta1  = (const float*)d_in[9];
    const float* Wa     = (const float*)d_in[10];
    const float* ba     = (const float*)d_in[11];
    const float* Wb     = (const float*)d_in[12];
    const float* bb     = (const float*)d_in[13];
    const float* gamma2 = (const float*)d_in[14];
    const float* beta2  = (const float*)d_in[15];

    const int N = N_NODES;
    const int E = in_sizes[1] / 2;
    const int* src = ei;
    const int* dst = ei + E;

    // ---- workspace layout (64B-aligned slabs) ----
    char* base = (char*)d_ws;
    size_t off = 0;
    auto alloc = [&](size_t bytes) -> char* {
        char* p = base + off;
        off += (bytes + 63) & ~(size_t)63;
        return p;
    };
    // zero-initialized region (atomic accumulators) first, one memset
    float* deg    = (float*)alloc((size_t)N * 4);
    float* x1     = (float*)alloc((size_t)N * E_DIM * 4);
    float* x2     = (float*)alloc((size_t)N * E_DIM * 4);
    float* stats1 = (float*)alloc(2 * C3 * 4);
    float* stats2 = (float*)alloc(2 * E_DIM * 4);
    size_t zero_bytes = off;
    float*    dinv = (float*)alloc((size_t)N * 4);
    float*    ss1  = (float*)alloc(2 * C3 * 4);
    float*    ss2  = (float*)alloc(2 * E_DIM * 4);
    float*    h    = (float*)alloc((size_t)N * C3 * 4);        // [N,384] pre-BN
    _Float16* xh   = (_Float16*)alloc((size_t)N * E_DIM * 2);  // f16 activations
    _Float16* x1h  = (_Float16*)alloc((size_t)N * E_DIM * 2);
    _Float16* x2h  = (_Float16*)alloc((size_t)N * E_DIM * 2);
    _Float16* hh   = (_Float16*)alloc((size_t)N * C3 * 2);     // BN1(h) in f16
    _Float16* mid  = (_Float16*)alloc((size_t)N * H_DIM * 2);  // FFN hidden
    _Float16* W0t  = (_Float16*)alloc((size_t)E_DIM * E_DIM * 2);
    _Float16* W1t  = (_Float16*)alloc((size_t)E_DIM * E_DIM * 2);
    _Float16* W2t  = (_Float16*)alloc((size_t)E_DIM * E_DIM * 2);
    _Float16* Wat  = (_Float16*)alloc((size_t)C3 * H_DIM * 2);
    _Float16* Wbt  = (_Float16*)alloc((size_t)H_DIM * E_DIM * 2);

    hipMemsetAsync(d_ws, 0, zero_bytes, stream);

    // ---- weight staging: f32 [K][N] -> f16 [N][K] ----
    transpose_f16<<<(E_DIM * E_DIM + 255) / 256, 256, 0, stream>>>(W0, W0t, E_DIM, E_DIM);
    transpose_f16<<<(E_DIM * E_DIM + 255) / 256, 256, 0, stream>>>(W1, W1t, E_DIM, E_DIM);
    transpose_f16<<<(E_DIM * E_DIM + 255) / 256, 256, 0, stream>>>(W2, W2t, E_DIM, E_DIM);
    transpose_f16<<<(C3 * H_DIM + 255) / 256, 256, 0, stream>>>(Wa, Wat, C3, H_DIM);
    transpose_f16<<<(H_DIM * E_DIM + 255) / 256, 256, 0, stream>>>(Wb, Wbt, H_DIM, E_DIM);

    // ---- degree / d^-1/2 ----
    deg_kernel<<<(E + 255) / 256, 256, 0, stream>>>(dst, deg, E);
    dinv_kernel<<<(N + 255) / 256, 256, 0, stream>>>(deg, dinv, N);

    // ---- two propagation hops + f16 staging of activations ----
    int prop_blocks = (int)(((size_t)E * 32 + 255) / 256);
    size_t n4 = (size_t)N * E_DIM / 4;
    int cvt_blocks = (int)((n4 + 255) / 256);
    cvt_f16x4<<<cvt_blocks, 256, 0, stream>>>(x, xh, n4);
    prop_kernel<<<prop_blocks, 256, 0, stream>>>(src, dst, dinv, x, x1, E);
    cvt_f16x4<<<cvt_blocks, 256, 0, stream>>>(x1, x1h, n4);
    prop_kernel<<<prop_blocks, 256, 0, stream>>>(src, dst, dinv, x1, x2, E);
    cvt_f16x4<<<cvt_blocks, 256, 0, stream>>>(x2, x2h, n4);

    // ---- per-hop 128x128 GEMMs into h[:, hop*128 : hop*128+128] ----
    int tiles_m = N / 16;                             // 3125
    {
        int tn = E_DIM / 16;                          // 8
        int blocks = (tiles_m * tn * 32 + 255) / 256; // 8 waves/block
        wmma_gemm<false, false, false><<<blocks, 256, 0, stream>>>(
            xh,  E_DIM, W0t, E_DIM, b0, nullptr, 0, h, C3, 0 * E_DIM, N, E_DIM, tn);
        wmma_gemm<false, false, false><<<blocks, 256, 0, stream>>>(
            x1h, E_DIM, W1t, E_DIM, b1, nullptr, 0, h, C3, 1 * E_DIM, N, E_DIM, tn);
        wmma_gemm<false, false, false><<<blocks, 256, 0, stream>>>(
            x2h, E_DIM, W2t, E_DIM, b2, nullptr, 0, h, C3, 2 * E_DIM, N, E_DIM, tn);
    }

    // ---- BN(384): stats on f32 h, apply fused with f16 conversion -> hh ----
    bn_stats<<<512, C3, 0, stream>>>(h, N, C3, stats1);
    bn_finalize<<<(C3 + 255) / 256, 256, 0, stream>>>(stats1, gamma1, beta1, ss1, C3, 1.0f / N);
    {
        size_t total = (size_t)N * C3;
        bn_apply_f16<<<(int)((total + 255) / 256), 256, 0, stream>>>(h, ss1, hh, C3, total);
    }

    // ---- FFN: mid = relu(hh @ Wa + ba) (f16),  d_out = mid @ Wb + bb + hh[:, :128]
    {
        int tn = H_DIM / 16;                          // 32
        int blocks = (tiles_m * tn * 32 + 255) / 256;
        wmma_gemm<true, true, false><<<blocks, 256, 0, stream>>>(
            hh, C3, Wat, C3, ba, nullptr, 0, mid, H_DIM, 0, N, C3, tn);
    }
    {
        int tn = E_DIM / 16;                          // 8
        int blocks = (tiles_m * tn * 32 + 255) / 256;
        wmma_gemm<false, false, true><<<blocks, 256, 0, stream>>>(
            mid, H_DIM, Wbt, H_DIM, bb, hh, C3, d_out, E_DIM, 0, N, H_DIM, tn);
    }

    // ---- BN(128) on d_out, in place ----
    float* outf = (float*)d_out;
    bn_stats<<<512, E_DIM, 0, stream>>>(outf, N, E_DIM, stats2);
    bn_finalize<<<1, 256, 0, stream>>>(stats2, gamma2, beta2, ss2, E_DIM, 1.0f / N);
    {
        size_t total = (size_t)N * E_DIM;
        bn_apply<<<(int)((total + 255) / 256), 256, 0, stream>>>(outf, ss2, E_DIM, total);
    }
}
